// PostProcess_26121991094504
// MI455X (gfx1250) — 8-bit matrix operand order repaired and checked by reading, compiled, not run on hardware
//
#include <hip/hip_runtime.h>
#include <stdint.h>


typedef __attribute__((ext_vector_type(8))) int v8i;

#define GRID_N    16777216
#define CHUNKS    (GRID_N / 64)
#define CNT_COLS  16
#define STATS_INTS (33*CNT_COLS + 33 + 35)
#define PANO_OFF  (33*CNT_COLS)
#define SEML_OFF  (33*CNT_COLS + 33)
#define GRID0_OFF 4096

__device__ __forceinline__ int eq4(int4 w, int t) {
  return (w.x == t ? 1 : 0) | (w.y == t ? (1 << 8) : 0) |
         (w.z == t ? (1 << 16) : 0) | (w.w == t ? (1 << 24) : 0);
}

__global__ void zero_stats(int* wsI) {
  for (int i = threadIdx.x; i < STATS_INTS; i += blockDim.x) wsI[i] = 0;
}

__global__ void counts_wmma(const int* __restrict__ inst,
                            const int* __restrict__ sem,
                            int* __restrict__ counts) {
  const int tid    = blockIdx.x * blockDim.x + threadIdx.x;
  const int lane   = threadIdx.x & 31;
  const int wave   = tid >> 5;
  const int nwaves = (gridDim.x * blockDim.x) >> 5;
  const int m      = lane & 15;
  const bool hi    = lane >= 16;

  v8i c1 = {0,0,0,0,0,0,0,0};
  v8i c2 = {0,0,0,0,0,0,0,0};

  for (int chunk = wave; chunk < CHUNKS; chunk += nwaves) {
    const int base = chunk << 6;
    v8i A1, A2, B;
#pragma unroll
    for (int j = 0; j < 8; ++j) {
      const int kbA = ((j & 1) << 2) + ((j & 2) << 3) + ((j >> 2) << 5) + (hi ? 8 : 0);
      int4 w = *reinterpret_cast<const int4*>(inst + base + kbA);
      A1[j] = eq4(w, 1 + m);
      A2[j] = eq4(w, 17 + m);
    }
#pragma unroll
    for (int j = 0; j < 8; ++j) {
      const int kbB = ((j & 1) << 2) + ((j & 2) << 3) + ((j >> 2) << 5) + (hi ? 8 : 0);
      int4 w = *reinterpret_cast<const int4*>(sem + base + kbB);
      B[j] = eq4(w, m);
    }
    c1 = __builtin_amdgcn_wmma_i32_16x16x64_iu8(false, A1, false, B, c1, false, false);
    c2 = __builtin_amdgcn_wmma_i32_16x16x64_iu8(false, A2, false, B, c2, false, false);
  }

#pragma unroll
  for (int j = 0; j < 8; ++j) {
    const int row = hi ? (j + 8) : j;
    atomicAdd(&counts[(1  + row) * CNT_COLS + m], c1[j]);
    atomicAdd(&counts[(17 + row) * CNT_COLS + m], c2[j]);
  }
}

__global__ void build_luts(const int* __restrict__ ids2d, int* wsI) {
  if (threadIdx.x != 0 || blockIdx.x != 0) return;
  int* counts = wsI;
  int* pano   = wsI + PANO_OFF;
  int* semlut = wsI + SEML_OFF;

  bool in2d[33];
  int  hist[33];
  int  pid[33];
  for (int i = 0; i < 33; ++i) in2d[i] = false;
  for (int t = 0; t < 16; ++t) {
    int v = ids2d[t] + 1;
    if (v >= 0 && v <= 32) in2d[v] = true;
  }
  long long tot = 0;
  for (int id = 1; id <= 32; ++id) {
    int s = 0;
    for (int n = 0; n < 13; ++n) s += counts[id * CNT_COLS + n];
    hist[id] = s;
    tot += s;
  }
  int zero_present = ((long long)GRID_N - tot) > 0 ? 1 : 0;
  for (int id = 1; id <= 32; ++id)
    if (hist[id] > 0 && !in2d[id]) zero_present = 1;

  pano[0] = 0;
  int rank = 0;
  for (int id = 1; id <= 32; ++id) {
    bool pres = in2d[id] && hist[id] > 0;
    if (pres) rank++;
    pid[id] = pres ? (rank - 1 + zero_present + 2) : 0;
    pano[id] = pid[id];
  }
  int sel[33];
  for (int id = 1; id <= 32; ++id) {
    int best = INT_MIN, bi = 0;
    for (int n = 0; n < 13; ++n) {
      int c = (n == 0 || n == 10 || n == 11) ? -1 : counts[id * CNT_COLS + n];
      if (c > best) { best = c; bi = n; }
    }
    sel[id] = bi;
  }
  for (int i = 0; i < 35; ++i) semlut[i] = 0;
  semlut[1] = 10;
  semlut[2] = 11;
  for (int id = 1; id <= 32; ++id) {
    bool pres = in2d[id] && hist[id] > 0;
    semlut[pid[id]] = pres ? sel[id] : 0;
  }
}

__global__ void grid0_build(const int* __restrict__ inst,
                            const int* __restrict__ sem,
                            const int* __restrict__ pano_lut,
                            uint8_t* __restrict__ grid0) {
  const int idx = blockIdx.x * blockDim.x + threadIdx.x;
  if (idx >= GRID_N) return;
  int p = pano_lut[inst[idx] & 31];
  const int s = sem[idx];
  if (s == 10) p = 1;
  else if (s == 11) p = 2;
  grid0[idx] = (uint8_t)p;
}

__global__ void nn_search(const float* __restrict__ geom,
                          const uint8_t* __restrict__ grid0,
                          const int* __restrict__ semlut,
                          float* __restrict__ out) {
  const int idx = blockIdx.x * blockDim.x + threadIdx.x;
  if (idx >= GRID_N) return;
  const int x = idx >> 16, y = (idx >> 8) & 255, z = idx & 255;
  int v = grid0[idx];
  if (v == 0 && fabsf(geom[idx]) <= 1.5f) {
    __builtin_prefetch(grid0 + ((((x + 253) & 255) << 16) |
                                (((y + 253) & 255) << 8)  |
                                ((z + 253) & 255)), 0, 0);
    int res = 0;
    for (int dx = -3; dx < 3 && !res; ++dx) {
      const int px = ((x + dx) & 255) << 16;
      for (int dy = -3; dy < 3 && !res; ++dy) {
        const int pxy = px | (((y + dy) & 255) << 8);
        for (int dz = -3; dz < 3; ++dz) {
          const int nb = grid0[pxy | ((z + dz) & 255)];
          if (nb > 0) { res = nb; break; }
        }
      }
    }
    v = res;
  }
  out[idx] = (float)v;
  out[GRID_N + idx] = (float)semlut[v];
}

extern "C" void kernel_launch(void* const* d_in, const int* in_sizes, int n_in,
                              void* d_out, int out_size, void* d_ws, size_t ws_size,
                              hipStream_t stream) {
  const float* geom  = (const float*)d_in[0];
  const int*   inst  = (const int*)d_in[1];
  const int*   sem   = (const int*)d_in[2];
  const int*   ids2d = (const int*)d_in[3];
  int*     wsI   = (int*)d_ws;
  uint8_t* grid0 = (uint8_t*)d_ws + GRID0_OFF;
  float*   out   = (float*)d_out;

  zero_stats <<<1,   256, 0, stream>>>(wsI);
  counts_wmma<<<256, 256, 0, stream>>>(inst, sem, wsI);
  build_luts <<<1,   64,  0, stream>>>(ids2d, wsI);
  grid0_build<<<GRID_N/256, 256, 0, stream>>>(inst, sem, wsI + PANO_OFF, grid0);
  nn_search  <<<GRID_N/256, 256, 0, stream>>>(geom, grid0, wsI + SEML_OFF, out);
}
